// GraphButterflyNetwork_5987184410999
// MI455X (gfx1250) — matrix operand order repaired and checked by reading, not compiled, not run on hardware
//
#include <hip/hip_runtime.h>

// ---------- types ----------
typedef __attribute__((ext_vector_type(16))) __bf16 v16bf;
typedef __attribute__((ext_vector_type(8)))  __bf16 v8bf;
typedef __attribute__((ext_vector_type(8)))  float  v8f;

#define BATCH      65536
#define ROWS_BLK   128
#define NBLK       (BATCH / ROWS_BLK)     // 512
#define DH         256
#define DOUT       128
#define LDS_STRIDE 272                     // 256 + 16 bf16 pad (row = 544 B, 16B-aligned)
#define EPS        1e-5f
#define NOISE_STD  0.1f

// ---------- weight prep: fp32 [k][n] -> bf16 transposed [n][k] ----------
__global__ __launch_bounds__(256) void prep_weights(const float* __restrict__ W1,
                                                    const float* __restrict__ W2,
                                                    __bf16* __restrict__ w1t,
                                                    __bf16* __restrict__ w2t) {
    int idx = blockIdx.x * 256 + threadIdx.x;
    if (idx < 5 * 256 * 256) {
        int nd = idx >> 16, r = idx & 65535, n = r >> 8, k = r & 255;
        w1t[idx] = (__bf16)W1[nd * 65536 + k * 256 + n];
    } else {
        int j = idx - 5 * 256 * 256;
        if (j < 5 * 128 * 256) {
            int nd = j >> 15, r = j & 32767, n = r >> 8, k = r & 255;
            w2t[j] = (__bf16)W2[nd * 32768 + k * 128 + n];
        }
    }
}

// ---------- fused 2-layer GEMM for one node ----------
// pA: cols [0,128), pB: cols [128,256), both row stride `stride` (handles concat).
__global__ __launch_bounds__(256) void node_gemm(const float* __restrict__ pA, int stride,
                                                 const float* __restrict__ pB,
                                                 const __bf16* __restrict__ w1t,  // [256][256]
                                                 const __bf16* __restrict__ w2t,  // [128][256]
                                                 const float* __restrict__ b1,    // [256]
                                                 const float* __restrict__ b2,    // [128]
                                                 float* __restrict__ y,           // [B][128]
                                                 float* __restrict__ partials) {  // [NBLK][256]
    __shared__ __align__(16) __bf16 xs[ROWS_BLK * LDS_STRIDE];
    __shared__ __align__(16) __bf16 hs[ROWS_BLK * LDS_STRIDE];
    __shared__ float sred[DOUT * 16];
    __shared__ float s2red[DOUT * 16];

    const int tid  = threadIdx.x;
    const int lane = tid & 31;
    const int wave = tid >> 5;
    const int l16  = lane & 15;
    const int hi   = lane >> 4;
    const size_t r0 = (size_t)blockIdx.x * ROWS_BLK;

    // ---- cooperative load of A tile (128 x 256 fp32) -> bf16 in LDS ----
    #pragma unroll
    for (int j = 0; j < 32; ++j) {
        int idx = j * 256 + tid;
        int row = idx >> 6;
        int col = (idx & 63) * 4;
        float4 v;
        if (col < 128) v = *(const float4*)(pA + (r0 + row) * (size_t)stride + col);
        else           v = *(const float4*)(pB + (r0 + row) * (size_t)stride + (col - 128));
        __bf16* d = &xs[row * LDS_STRIDE + col];
        d[0] = (__bf16)v.x; d[1] = (__bf16)v.y; d[2] = (__bf16)v.z; d[3] = (__bf16)v.w;
    }
    __syncthreads();

    const int rowA    = wave * 16 + l16;       // A-fragment row for this lane
    const int kb      = hi * 16;               // A: lane>=16 holds K+16 block
    const int kb2     = hi * 16;               // B: lane>=16 holds K+16 block
    const int rowbase = wave * 16 + hi * 8;    // C/D rows: i + 8*(lane>=16)

    // ---- GEMM1: h = relu(x @ W1 + b1), keep all A fragments in VGPRs ----
    v16bf afrag[8];
    #pragma unroll
    for (int kt = 0; kt < 8; ++kt) {
        const __bf16* p = &xs[rowA * LDS_STRIDE + kt * 32 + kb];
        v8bf lo = *(const v8bf*)p;
        v8bf hv = *(const v8bf*)(p + 8);
        afrag[kt] = __builtin_shufflevector(lo, hv, 0,1,2,3,4,5,6,7,8,9,10,11,12,13,14,15);
    }
    for (int nt = 0; nt < 16; ++nt) {
        const int n = nt * 16 + l16;
        const float bias = b1[n];
        v8f acc = {bias, bias, bias, bias, bias, bias, bias, bias};
        #pragma unroll
        for (int kt = 0; kt < 8; ++kt) {
            v16bf bfrag = *(const v16bf*)(w1t + (size_t)n * 256 + kt * 32 + kb2);
            acc = __builtin_amdgcn_wmma_f32_16x16x32_bf16(false, afrag[kt], false, bfrag,
                                                          (short)0, acc, false, false);
        }
        #pragma unroll
        for (int i = 0; i < 8; ++i) {
            float v = acc[i];
            v = v > 0.f ? v : 0.f;
            hs[(rowbase + i) * LDS_STRIDE + n] = (__bf16)v;
        }
    }

    // ---- GEMM2: y = h @ W2 + b2 (wave consumes only its own hs rows: no barrier) ----
    v16bf hfrag[8];
    #pragma unroll
    for (int kt = 0; kt < 8; ++kt) {
        const __bf16* p = &hs[rowA * LDS_STRIDE + kt * 32 + kb];
        v8bf lo = *(const v8bf*)p;
        v8bf hv = *(const v8bf*)(p + 8);
        hfrag[kt] = __builtin_shufflevector(lo, hv, 0,1,2,3,4,5,6,7,8,9,10,11,12,13,14,15);
    }
    for (int nt = 0; nt < 8; ++nt) {
        const int n = nt * 16 + l16;
        const float bias = b2[n];
        v8f acc = {bias, bias, bias, bias, bias, bias, bias, bias};
        #pragma unroll
        for (int kt = 0; kt < 8; ++kt) {
            v16bf bfrag = *(const v16bf*)(w2t + (size_t)n * 256 + kt * 32 + kb2);
            acc = __builtin_amdgcn_wmma_f32_16x16x32_bf16(false, hfrag[kt], false, bfrag,
                                                          (short)0, acc, false, false);
        }
        float s = 0.f, s2 = 0.f;
        #pragma unroll
        for (int i = 0; i < 8; ++i) {
            float v = acc[i];
            y[(r0 + rowbase + i) * DOUT + n] = v;
            s += v; s2 += v * v;
        }
        sred[n * 16 + wave * 2 + hi]  = s;   // fixed slot per (wave, half) -> deterministic
        s2red[n * 16 + wave * 2 + hi] = s2;
    }
    __syncthreads();

    // ---- deterministic block-level reduction of per-channel stats ----
    if (tid < DOUT) {
        float s = 0.f, s2 = 0.f;
        #pragma unroll
        for (int j = 0; j < 16; ++j) { s += sred[tid * 16 + j]; s2 += s2red[tid * 16 + j]; }
        partials[(size_t)blockIdx.x * 256 + tid]        = s;
        partials[(size_t)blockIdx.x * 256 + 128 + tid]  = s2;
    }
}

// ---------- BN stats finalize: mu / rsqrt(var+eps), deterministic order ----------
__global__ __launch_bounds__(128) void bn_finalize(const float* __restrict__ partials,
                                                   float* __restrict__ murstd) {
    int t = threadIdx.x;  // 0..127
    float s = 0.f, s2 = 0.f;
    for (int b = 0; b < NBLK; ++b) {
        s  += partials[(size_t)b * 256 + t];
        s2 += partials[(size_t)b * 256 + 128 + t];
    }
    float mu  = s * (1.f / (float)BATCH);
    float var = s2 * (1.f / (float)BATCH) - mu * mu;
    murstd[t]        = mu;
    murstd[128 + t]  = rsqrtf(var + EPS);
}

// ---------- normalize (+ optional noise), writes final output region ----------
__global__ __launch_bounds__(256) void bn_apply(const float* __restrict__ y,
                                                const float* __restrict__ murstd,
                                                const float* __restrict__ gamma,
                                                const float* __restrict__ beta,
                                                const float* __restrict__ noise,
                                                float* __restrict__ outp) {
    size_t i = (size_t)blockIdx.x * 256 + threadIdx.x;   // over B*128/4 float4s
    float4 v = ((const float4*)y)[i];
    int c0 = ((int)i & 31) * 4;
    float r[4] = {v.x, v.y, v.z, v.w};
    #pragma unroll
    for (int j = 0; j < 4; ++j) {
        int c = c0 + j;
        r[j] = gamma[c] * (r[j] - murstd[c]) * murstd[128 + c] + beta[c];
    }
    if (noise) {
        float4 nz = ((const float4*)noise)[i];
        r[0] += NOISE_STD * nz.x; r[1] += NOISE_STD * nz.y;
        r[2] += NOISE_STD * nz.z; r[3] += NOISE_STD * nz.w;
    }
    float4 o = {r[0], r[1], r[2], r[3]};
    ((float4*)outp)[i] = o;
}

// ---------- host orchestration ----------
extern "C" void kernel_launch(void* const* d_in, const int* in_sizes, int n_in,
                              void* d_out, int out_size, void* d_ws, size_t ws_size,
                              hipStream_t stream) {
    const float* x1    = (const float*)d_in[0];
    const float* x2    = (const float*)d_in[1];
    const float* W1    = (const float*)d_in[2];
    const float* b1    = (const float*)d_in[3];
    const float* W2    = (const float*)d_in[4];
    const float* b2    = (const float*)d_in[5];
    const float* gamma = (const float*)d_in[6];
    const float* beta  = (const float*)d_in[7];
    const float* noise = (const float*)d_in[8];
    float* out = (float*)d_out;

    char* ws = (char*)d_ws;
    float*  y        = (float*)(ws);                 // B*128 f32      = 33,554,432 B
    float*  partials = (float*)(ws + 33554432);      // 512*256 f32    =    524,288 B
    float*  murstd   = (float*)(ws + 34078720);      // 256 f32        =      1,024 B
    __bf16* w1t      = (__bf16*)(ws + 34079744);     // 5*256*256 bf16 =    655,360 B
    __bf16* w2t      = (__bf16*)(ws + 34735104);     // 5*128*256 bf16 =    327,680 B

    prep_weights<<<1920, 256, 0, stream>>>(W1, W2, w1t, w2t);

    const size_t S = (size_t)BATCH * DOUT;
    // outputs concatenated in return order (o3, o4, o0, o1, o2)
    float* o0 = out + 2 * S; float* o1 = out + 3 * S; float* o2 = out + 4 * S;
    float* o3 = out;         float* o4 = out + S;

    const float* nA[5]  = { x1,       x2,       o0, o0, o1 };
    const float* nB[5]  = { x1 + 128, x2 + 128, o1, o2, o2 };
    const int    st[5]  = { 256, 256, 128, 128, 128 };
    float*       op[5]  = { o0, o1, o2, o3, o4 };
    const float* nz[5]  = { nullptr, nullptr, noise, noise + S, noise + 2 * S };

    for (int nd = 0; nd < 5; ++nd) {
        node_gemm<<<NBLK, 256, 0, stream>>>(nA[nd], st[nd], nB[nd],
                                            w1t + (size_t)nd * 65536,
                                            w2t + (size_t)nd * 32768,
                                            b1 + nd * 256, b2 + nd * 128,
                                            y, partials);
        bn_finalize<<<1, 128, 0, stream>>>(partials, murstd);
        bn_apply<<<(BATCH * DOUT / 4) / 256, 256, 0, stream>>>(y, murstd,
                                                               gamma + nd * 128,
                                                               beta + nd * 128,
                                                               nz[nd], op[nd]);
    }
}
